// GatedRelevanceNetwork_27307402067994
// MI455X (gfx1250) — compile-verified, matrix operand order checked by reading, not run on hardware
//
#include <hip/hip_runtime.h>
#include <hip/hip_bf16.h>
#include <math.h>

// Problem constants (from reference): B=32, L1=L2=128, D=300, KDIM=50
#define BB   32
#define LL   128
#define DD   300
#define DP   320      // D padded to 10 * 32 for wmma K-steps
#define KD   50
#define KG   5        // k's per workgroup
#define NKG  (KD / KG)

typedef __attribute__((ext_vector_type(16))) _Float16 v16h;
typedef __attribute__((ext_vector_type(8)))  _Float16 v8h;
typedef __attribute__((ext_vector_type(8)))  float    v8f;

// ---------------- fragment helpers ----------------------------------------
// CDNA5 16-bit A/B 16x16x32 lane layout:
//   lanes 0-15  hold (row/col = lane)    K = {o..o+7} U {o+16..o+23} with o = kbase
//   lanes 16-31 hold (row/col = lane-16) K shifted by +8
// Each lane therefore loads two contiguous 16-byte chunks from its row.
template <typename P>
__device__ __forceinline__ v16h ldfrag(P row, int o) {
    v8h lo = *(const v8h*)(row + o);
    v8h hi = *(const v8h*)(row + o + 16);
    v16h r;
#pragma unroll
    for (int t = 0; t < 8; ++t) { r[t] = lo[t]; r[t + 8] = hi[t]; }
    return r;
}

__device__ __forceinline__ v8f wmma16(v16h a, v16h b, v8f c) {
    return __builtin_amdgcn_wmma_f32_16x16x32_f16(
        /*neg_a=*/false, a, /*neg_b=*/false, b,
        /*c_mod=*/(short)0, c, /*reuse_a=*/false, /*reuse_b=*/false);
}

// ---------------- prep kernels --------------------------------------------
__global__ void cvt_rows_pad(const float* __restrict__ src,
                             _Float16* __restrict__ dst, int nrows) {
    int idx = blockIdx.x * blockDim.x + threadIdx.x;
    int total = nrows * DP;
    if (idx >= total) return;
    int dp = idx % DP, row = idx / DP;
    float v = (dp < DD) ? src[(size_t)row * DD + dp] : 0.f;
    dst[idx] = (_Float16)v;
}

// wbh[k][e][d] = Wb[k, d, e]   (transposed so B-fragments read contiguous d)
__global__ void cvt_wb_t(const float* __restrict__ Wb, _Float16* __restrict__ dst) {
    int idx = blockIdx.x * blockDim.x + threadIdx.x;
    int total = KD * DP * DP;
    if (idx >= total) return;
    int d = idx % DP;
    int t = idx / DP;
    int e = t % DP;
    int k = t / DP;
    float v = (d < DD && e < DD) ? Wb[((size_t)k * DD + d) * DD + e] : 0.f;
    dst[idx] = (_Float16)v;
}

// pd[row,k] = sum_d E[row,d]*Wdp[d,k];  pg likewise with Wgp.  row = b*LL+i
__global__ void proj_kernel(const float* __restrict__ E,
                            const float* __restrict__ Wdp,
                            const float* __restrict__ Wgp,
                            float* __restrict__ pd, float* __restrict__ pg) {
    int idx = blockIdx.x * blockDim.x + threadIdx.x;
    if (idx >= BB * LL * KD) return;
    int k = idx % KD, row = idx / KD;
    const float* ev = E + (size_t)row * DD;
    float sd = 0.f, sg = 0.f;
    for (int d = 0; d < DD; ++d) {
        float e = ev[d];
        sd = fmaf(e, Wdp[d * KD + k], sd);
        sg = fmaf(e, Wgp[d * KD + k], sg);
    }
    pd[idx] = sd;
    pg[idx] = sg;
}

__global__ void zero_kernel(float* __restrict__ p, int n) {
    int i = blockIdx.x * blockDim.x + threadIdx.x;
    if (i < n) p[i] = 0.f;
}

// ---------------- fused bilinear + gating kernel --------------------------
// grid = (NKG, BB), block = 256 (8 waves).  Each wave owns M-rows [w*16, w*16+16).
#define TROW 72   // LDS row stride in halfs (144B: 16B-aligned, bank-conflict pad)

__global__ __launch_bounds__(256, 1)
__attribute__((amdgpu_waves_per_eu(4)))   // cap ~256 VGPRs/wave: no vgpr-msb traffic
void fused_kernel(const _Float16* __restrict__ e1h,   // [B][LL][DP]
                  const _Float16* __restrict__ e2h,   // [B][LL][DP]
                  const _Float16* __restrict__ wbh,   // [KD][DP][DP] (e,d)
                  const float* __restrict__ p1d, const float* __restrict__ p2d,
                  const float* __restrict__ p1g, const float* __restrict__ p2g,
                  const float* __restrict__ bg, const float* __restrict__ bvec,
                  const float* __restrict__ u, float* __restrict__ out) {
    const int kg   = blockIdx.x;           // k-group 0..9
    const int b    = blockIdx.y;           // batch
    const int tid  = threadIdx.x;
    const int wave = tid >> 5;             // 0..7
    const int lane = tid & 31;
    const int ln   = lane & 15;
    const int hi   = lane >> 4;            // 0: K{0..7,16..23}; 1: K{8..15,24..31}
    const int m0   = wave * 16;

    __shared__ _Float16 Tl[LL][TROW];      // 18 KB, per-wave-private row slabs
    __shared__ float oaccL[64][256];       // 64 KB, k-group output accumulator
                                           // [jb*8+r][tid]: thread-private column

    const _Float16* Arow   = e1h + ((size_t)(b * LL + m0 + ln)) * DP;
    const _Float16* e2base = e2h + (size_t)b * LL * DP;
    const _Float16* TlA    = &Tl[m0 + ln][0];

    // zero own accumulator column (no barrier needed: private slots)
#pragma unroll
    for (int s = 0; s < 64; ++s) oaccL[s][tid] = 0.f;

    v8f zero = {0.f, 0.f, 0.f, 0.f, 0.f, 0.f, 0.f, 0.f};

    for (int kt = 0; kt < KG; ++kt) {
        const int k = kg * KG + kt;
        const _Float16* wrow = wbh + (size_t)k * DP * DP;

        // prefetch next k's 200KB Wb slab into L2/WGP$ while computing this one
        if (kt + 1 < KG) {
            const char* nxt = (const char*)(wrow + (size_t)DP * DP) + tid * 800;
#pragma unroll
            for (int p = 0; p < 800; p += 128)
                __builtin_prefetch(nxt + p, 0, 0);
        }

        v8f c2[8];
#pragma unroll
        for (int jb = 0; jb < 8; ++jb) c2[jb] = zero;

        // e-chunks of 64: GEMM1 -> T slab in LDS, then GEMM2 partial contraction
        for (int ec = 0; ec < DP / 64; ++ec) {
            const int e0c = ec * 64;
            // --- GEMM1: T[m0..m0+15, e0c..e0c+63] = e1 @ Wb[k] --------------
#pragma unroll
            for (int nb = 0; nb < 4; ++nb) {
                const _Float16* Brow = wrow + (size_t)(e0c + nb * 16 + ln) * DP;
                v8f acc = zero;
#pragma unroll
                for (int kb = 0; kb < DP / 32; ++kb) {
                    v16h a  = ldfrag(Arow, kb * 32 + hi * 8);
                    v16h bf = ldfrag(Brow, kb * 32 + hi * 8);
                    acc = wmma16(a, bf, acc);
                }
                // C layout: VGPR r -> M = r + 8*hi, N = ln
#pragma unroll
                for (int r = 0; r < 8; ++r)
                    Tl[m0 + r + hi * 8][nb * 16 + ln] = (_Float16)acc[r];
            }
            // same-wave LDS RAW: DS ops are in-order per wave (DScnt) — no barrier
            // --- GEMM2 partial: c2 += T_chunk @ e2^T ------------------------
#pragma unroll
            for (int jb = 0; jb < 8; ++jb) {
                const _Float16* Bj = e2base + (size_t)(jb * 16 + ln) * DP;
#pragma unroll
                for (int kk = 0; kk < 2; ++kk) {
                    v16h a  = ldfrag(TlA, kk * 32 + hi * 8);
                    v16h bf = ldfrag(Bj, e0c + kk * 32 + hi * 8);
                    c2[jb] = wmma16(a, bf, c2[jb]);
                }
            }
        }

        // --- gating epilogue for this k, accumulate mix*u[k] into LDS ------
        const float uk = u[k], bk = bvec[k], bgk = bg[k];
#pragma unroll
        for (int jb = 0; jb < 8; ++jb) {
            const int j = jb * 16 + ln;
            const float p2dv = p2d[(size_t)(b * LL + j) * KD + k];
            const float p2gv = p2g[(size_t)(b * LL + j) * KD + k];
#pragma unroll
            for (int r = 0; r < 8; ++r) {
                const int i = m0 + r + hi * 8;
                const float btp = c2[jb][r];
                const float sln = tanhf(p1d[(size_t)(b * LL + i) * KD + k] + p2dv);
                const float zg  = p1g[(size_t)(b * LL + i) * KD + k] + p2gv + bgk;
                const float g   = 1.f / (1.f + __expf(-zg));
                oaccL[jb * 8 + r][tid] += (g * btp + (1.f - g) * sln + bk) * uk;
            }
        }
    }

    // one atomic pass: out[b,i,j] += sum_{k in group} mix*u[k]
#pragma unroll
    for (int jb = 0; jb < 8; ++jb) {
        const int j = jb * 16 + ln;
#pragma unroll
        for (int r = 0; r < 8; ++r) {
            const int i = m0 + r + hi * 8;
            atomicAdd(&out[((size_t)b * LL + i) * LL + j], oaccL[jb * 8 + r][tid]);
        }
    }
}

// ---------------- launch ---------------------------------------------------
extern "C" void kernel_launch(void* const* d_in, const int* in_sizes, int n_in,
                              void* d_out, int out_size, void* d_ws, size_t ws_size,
                              hipStream_t stream) {
    const float* e1 = (const float*)d_in[0];
    const float* e2 = (const float*)d_in[1];
    const float* Wb = (const float*)d_in[2];
    const float* Wd = (const float*)d_in[3];
    const float* Wg = (const float*)d_in[4];
    const float* bg = (const float*)d_in[5];
    const float* bv = (const float*)d_in[6];
    const float* u  = (const float*)d_in[7];
    float* out = (float*)d_out;

    // workspace layout (bytes)
    char* ws = (char*)d_ws;
    const size_t sz_e = (size_t)BB * LL * DP * sizeof(_Float16);      // 2.62 MB each
    const size_t sz_w = (size_t)KD * DP * DP * sizeof(_Float16);      // 10.24 MB
    const size_t sz_p = (size_t)BB * LL * KD * sizeof(float);         // 0.82 MB each
    _Float16* e1h = (_Float16*)(ws);
    _Float16* e2h = (_Float16*)(ws + sz_e);
    _Float16* wbh = (_Float16*)(ws + 2 * sz_e);
    float* p1d = (float*)(ws + 2 * sz_e + sz_w);
    float* p2d = (float*)(ws + 2 * sz_e + sz_w + sz_p);
    float* p1g = (float*)(ws + 2 * sz_e + sz_w + 2 * sz_p);
    float* p2g = (float*)(ws + 2 * sz_e + sz_w + 3 * sz_p);

    const int T = 256;
    // f32 -> f16 (padded) conversions
    {
        int n = BB * LL * DP;
        cvt_rows_pad<<<(n + T - 1) / T, T, 0, stream>>>(e1, e1h, BB * LL);
        cvt_rows_pad<<<(n + T - 1) / T, T, 0, stream>>>(e2, e2h, BB * LL);
        int nw = KD * DP * DP;
        cvt_wb_t<<<(nw + T - 1) / T, T, 0, stream>>>(Wb, wbh);
    }
    // small projections p1d/p1g (from e1, Wd1/Wg1) and p2d/p2g (from e2, Wd2/Wg2)
    {
        int n = BB * LL * KD;
        proj_kernel<<<(n + T - 1) / T, T, 0, stream>>>(e1, Wd, Wg, p1d, p1g);
        proj_kernel<<<(n + T - 1) / T, T, 0, stream>>>(e2, Wd + DD * KD, Wg + DD * KD,
                                                       p2d, p2g);
    }
    // zero output (harness poisons it), then fused bilinear+gating
    zero_kernel<<<(out_size + T - 1) / T, T, 0, stream>>>(out, out_size);
    dim3 grid(NKG, BB);
    fused_kernel<<<grid, 256, 0, stream>>>(e1h, e2h, wbh, p1d, p2d, p1g, p2g,
                                           bg, bv, u, out);
}
